// sLSTMNeuron_71914932404784
// MI455X (gfx1250) — matrix operand order repaired and checked by reading, not compiled, not run on hardware
//
#include <hip/hip_runtime.h>
#include <hip/hip_bf16.h>
#include <math.h>

// ---------------- problem constants ----------------
#define NHEADS 8
#define HDIM   128
#define DDIM   1024
#define KCONV  4
#define BB     8
#define SS     2048
#define MM     (BB * SS)   // 16384 rows
#define NN     1024        // output cols of both big GEMMs
#define KK     1024        // reduction depth of both big GEMMs

typedef __attribute__((ext_vector_type(16))) __bf16 v16bf;
typedef __attribute__((ext_vector_type(8)))  float  v8f;
typedef __attribute__((ext_vector_type(4)))  unsigned v4u;
typedef __attribute__((ext_vector_type(8)))  int      v8i;
typedef __attribute__((ext_vector_type(4)))  int      v4i;

// ---------------- bf16 helpers (RNE) ----------------
__device__ __forceinline__ unsigned bf16rne(float f) {
  unsigned u = __float_as_uint(f);
  return (u + 0x7FFFu + ((u >> 16) & 1u)) >> 16;
}
__device__ __forceinline__ unsigned packbf2(float lo, float hi) {
  return bf16rne(lo) | (bf16rne(hi) << 16);
}
__device__ __forceinline__ float bfh2f(unsigned short h) {
  return __uint_as_float(((unsigned)h) << 16);
}

// =====================================================================
// Kernel 1: causal depthwise conv + SiLU + skinny gate projections.
// One block covers (b, 16 timesteps); 512 threads x 2 channels each.
// =====================================================================
#define GT    512
#define GCH   2
#define STILE 16

__global__ __launch_bounds__(GT) void gates_kernel(
    const float* __restrict__ x, const float* __restrict__ conv_w,
    const float* __restrict__ conv_b, const float* __restrict__ Wi,
    const float* __restrict__ Wf, const float* __restrict__ Wo,
    float* __restrict__ gI, float* __restrict__ gF, float* __restrict__ gO) {
  const int b  = blockIdx.x >> 7;            // SS/STILE = 128 tiles per batch
  const int s0 = (blockIdx.x & 127) * STILE;
  const int t  = threadIdx.x;

  int   d[GCH];
  float cw[GCH][KCONV], cb[GCH];
  float wi[GCH][NHEADS], wf[GCH][NHEADS], wo[GCH][NHEADS];
#pragma unroll
  for (int j = 0; j < GCH; ++j) {
    d[j]  = t + j * GT;
    cb[j] = conv_b[d[j]];
#pragma unroll
    for (int k = 0; k < KCONV; ++k) cw[j][k] = conv_w[d[j] * KCONV + k];
#pragma unroll
    for (int g = 0; g < NHEADS; ++g) {
      wi[j][g] = Wi[d[j] * NHEADS + g];
      wf[j][g] = Wf[d[j] * NHEADS + g];
      wo[j][g] = Wo[d[j] * NHEADS + g];
    }
  }

  float h0[GCH], h1[GCH], h2[GCH];
#pragma unroll
  for (int j = 0; j < GCH; ++j) {
    h0[j] = (s0 - 3 >= 0) ? x[((size_t)(b * SS + s0 - 3)) * DDIM + d[j]] : 0.f;
    h1[j] = (s0 - 2 >= 0) ? x[((size_t)(b * SS + s0 - 2)) * DDIM + d[j]] : 0.f;
    h2[j] = (s0 - 1 >= 0) ? x[((size_t)(b * SS + s0 - 1)) * DDIM + d[j]] : 0.f;
  }

  __shared__ float red[GT / 32][24];

  for (int si = 0; si < STILE; ++si) {
    const int s = s0 + si;
    const size_t row = (size_t)(b * SS + s);
    float acc[24];
#pragma unroll
    for (int g = 0; g < 24; ++g) acc[g] = 0.f;

#pragma unroll
    for (int j = 0; j < GCH; ++j) {
      float cur = x[row * DDIM + d[j]];
      float xc = cb[j] + h0[j] * cw[j][0] + h1[j] * cw[j][1] +
                 h2[j] * cw[j][2] + cur * cw[j][3];
      xc = xc / (1.f + expf(-xc));  // SiLU
#pragma unroll
      for (int g = 0; g < NHEADS; ++g) {
        acc[g]      += xc  * wi[j][g];
        acc[8 + g]  += xc  * wf[j][g];
        acc[16 + g] += cur * wo[j][g];
      }
      h0[j] = h1[j]; h1[j] = h2[j]; h2[j] = cur;
    }

#pragma unroll
    for (int off = 16; off > 0; off >>= 1)
#pragma unroll
      for (int g = 0; g < 24; ++g) acc[g] += __shfl_xor(acc[g], off, 32);
    if ((t & 31) == 0)
#pragma unroll
      for (int g = 0; g < 24; ++g) red[t >> 5][g] = acc[g];
    __syncthreads();
    if (t < 24) {
      float v = 0.f;
#pragma unroll
      for (int w = 0; w < GT / 32; ++w) v += red[w][t];
      float sc = 15.f * tanhf(v * (1.f / 15.f));  // softcap
      int head = t & 7;
      if (t < 8)       gI[row * NHEADS + head] = sc;
      else if (t < 16) gF[row * NHEADS + head] = sc;
      else             gO[row * NHEADS + head] = 1.f / (1.f + expf(-sc));
    }
    __syncthreads();
  }
}

// =====================================================================
// TDM: DMA a 128-row x 32-bf16 tile (row stride 1024 bf16) into LDS,
// padding each 64B row with 4 DWORDs so LDS row stride = 20 uints.
// D# layout per CDNA5 ISA.
// =====================================================================
__device__ __forceinline__ void tdm_load_tile_bf16(const void* gtile,
                                                   unsigned ldsByteAddr) {
  unsigned long long ga = (unsigned long long)gtile;
  v4u g0;
  g0[0] = 1u;                                     // count=1, user descriptor
  g0[1] = ldsByteAddr;                            // lds_addr
  g0[2] = (unsigned)(ga & 0xFFFFFFFFull);         // global_addr[31:0]
  g0[3] = (unsigned)((ga >> 32) & 0x1FFFFFFull)   // global_addr[56:32]
          | (2u << 30);                           // type=2 (image)
  v8i g1;
  g1[0] = (int)((1u << 16)      // data_size = 2 bytes
                | (1u << 20)    // pad_enable
                | (3u << 22)    // pad_interval: every 16 DWORDs (one row)
                | (3u << 25));  // pad_amount: 4 DWORDs
  g1[1] = (int)((1024u & 0xFFFFu) << 16);  // tensor_dim0 = 1024 (bits 63:48)
  g1[2] = 0;                               // dim0 hi, tensor_dim1 lo
  g1[3] = (int)(16u | (32u << 16));        // tensor_dim1 = 1<<20; tile_dim0 = 32
  g1[4] = (int)128u;                       // tile_dim1 = 128; tile_dim2 = 0
  g1[5] = (int)1024u;                      // tensor_dim0_stride = 1024 elems
  g1[6] = 0;
  g1[7] = 0;
  v4i z4 = {0, 0, 0, 0};
  v8i z8 = {0, 0, 0, 0, 0, 0, 0, 0};
  // 6-arg toolchain form: (g0, g1, g2, g3, g4, cpol)
  __builtin_amdgcn_tensor_load_to_lds(g0, g1, z4, z4, z8, 0);
}

// =====================================================================
// Kernel 2/4: bf16 WMMA GEMM, C = act(A @ W).
// Block tile 128x128, K-step 32, 8 waves, wave tile 64x32 (4x2 wmma).
// Software pipelined: next K-tile global loads issued during compute.
// fp32 A path: VALU convert->LDS.  bf16 A path: TDM tensor_load_to_lds.
// =====================================================================
#define BM 128
#define BN 128
#define BK 32
#define LDA_STRIDE 20  // uints per A row (16 data + 4 pad, TDM-compatible)

union FragBF { unsigned u[8]; v16bf v; };

template <bool IN_F32, bool TANH_BF16_OUT>
__global__ __launch_bounds__(256) void wmma_gemm(
    const void* __restrict__ Ain, const float* __restrict__ W,
    void* __restrict__ Out) {
  __shared__ __align__(16) unsigned ldsA[BM * LDA_STRIDE];  // [row][K-pair]
  __shared__ unsigned ldsB[(BK / 2) * BN];                  // [K-pair][n]

  const int t    = threadIdx.x;
  const int lane = t & 31;
  const int w    = t >> 5;
  const int wm   = w >> 2;  // 0..1  (64-row slabs)
  const int wn   = w & 3;   // 0..3  (32-col slabs)
  const int m0   = blockIdx.y * BM;
  const int n0   = blockIdx.x * BN;

  v8f acc[4][2];
#pragma unroll
  for (int mi = 0; mi < 4; ++mi)
#pragma unroll
    for (int ni = 0; ni < 2; ++ni)
#pragma unroll
      for (int r = 0; r < 8; ++r) acc[mi][ni][r] = 0.f;

  // ---- pipeline registers ----
  float  bw0[8], bw1[8];
  float2 apre[8];

  auto loadB = [&](int k0) {
#pragma unroll
    for (int r = 0; r < 8; ++r) {
      int slot = t + 256 * r;  // 16 pairs x 128 cols
      int p = slot >> 7, n = slot & 127;
      bw0[r] = W[((size_t)(k0 + 2 * p)) * NN + n0 + n];
      bw1[r] = W[((size_t)(k0 + 2 * p + 1)) * NN + n0 + n];
    }
  };
  auto loadA = [&](int k0) {
    const float2* A2 = (const float2*)Ain;
#pragma unroll
    for (int r = 0; r < 8; ++r) {
      int slot = t + 256 * r;  // 2048 uint slots
      int row = slot >> 4, cp = slot & 15;
      apre[r] = A2[((size_t)(m0 + row)) * (KK / 2) + (k0 >> 1) + cp];
    }
  };

  if (IN_F32) loadA(0);
  loadB(0);

  for (int k0 = 0; k0 < KK; k0 += BK) {
    __syncthreads();  // previous compute done reading LDS
    // ---- commit staged tile to LDS ----
    if (IN_F32) {
#pragma unroll
      for (int r = 0; r < 8; ++r) {
        int slot = t + 256 * r;
        int row = slot >> 4, cp = slot & 15;
        ldsA[row * LDA_STRIDE + cp] = packbf2(apre[r].x, apre[r].y);
      }
    } else if (w == 0) {
      // Tensor Data Mover: DMA bf16 tile with per-row LDS padding.
      const char* gtile =
          (const char*)Ain + ((size_t)m0 * KK + k0) * 2;
      tdm_load_tile_bf16(gtile, (unsigned)(unsigned long long)(uintptr_t)ldsA);
      __builtin_amdgcn_s_wait_tensorcnt(0);
    }
#pragma unroll
    for (int r = 0; r < 8; ++r) {
      int slot = t + 256 * r;
      int p = slot >> 7, n = slot & 127;
      ldsB[p * BN + n] = packbf2(bw0[r], bw1[r]);
    }
    __syncthreads();

    // ---- issue next tile's global loads (overlap with compute) ----
    int kn = k0 + BK;
    if (kn < KK) {
      if (IN_F32) loadA(kn);
      loadB(kn);
    }

    // ---- fragment loads per ISA lane layouts + WMMA ----
    FragBF af[4], bf[2];
#pragma unroll
    for (int mi = 0; mi < 4; ++mi) {
      int ml = wm * 64 + mi * 16 + (lane & 15);
      int base = (lane >> 4) * 4;
#pragma unroll
      for (int j = 0; j < 8; ++j)
        af[mi].u[j] = ldsA[ml * LDA_STRIDE + base + (j & 3) + ((j >> 2) << 3)];
    }
#pragma unroll
    for (int ni = 0; ni < 2; ++ni) {
      int nc = wn * 32 + ni * 16 + (lane & 15);
      int pb = (lane >> 4) << 2;
#pragma unroll
      for (int j = 0; j < 8; ++j)
        bf[ni].u[j] = ldsB[(pb + (j & 3) + ((j >> 2) << 3)) * BN + nc];
    }
#pragma unroll
    for (int mi = 0; mi < 4; ++mi)
#pragma unroll
      for (int ni = 0; ni < 2; ++ni)
        acc[mi][ni] = __builtin_amdgcn_wmma_f32_16x16x32_bf16(
            false, af[mi].v, false, bf[ni].v, (short)0, acc[mi][ni],
            false, false);
  }

  // ---- epilogue: C layout = VGPR r -> row r + 8*(lane>>4), col lane&15 ----
#pragma unroll
  for (int mi = 0; mi < 4; ++mi)
#pragma unroll
    for (int ni = 0; ni < 2; ++ni)
#pragma unroll
      for (int r = 0; r < 8; ++r) {
        int row = m0 + wm * 64 + mi * 16 + r + ((lane >> 4) << 3);
        int col = n0 + wn * 32 + ni * 16 + (lane & 15);
        float v = acc[mi][ni][r];
        if (TANH_BF16_OUT)
          ((unsigned short*)Out)[(size_t)row * NN + col] =
              (unsigned short)bf16rne(tanhf(v));
        else
          ((float*)Out)[(size_t)row * NN + col] = v;
      }
}

// =====================================================================
// Kernel 3: stabilized exponential-gating scan + fused per-head GroupNorm.
// One block per (b, head); 128 threads = H channels; c/n/m in registers.
// =====================================================================
__global__ __launch_bounds__(HDIM) void scan_kernel(
    const unsigned short* __restrict__ zbuf, const float* __restrict__ gI,
    const float* __restrict__ gF, const float* __restrict__ gO,
    const float* __restrict__ gamma, const float* __restrict__ beta,
    unsigned short* __restrict__ hnbuf, float* __restrict__ stateOut) {
  const int b    = blockIdx.x >> 3;
  const int head = blockIdx.x & 7;
  const int t    = threadIdx.x;
  const int wid  = t >> 5;
  __shared__ float red[8];

  float c = 0.f, n = 0.f, m = 0.f;
  const float gg = gamma[head * HDIM + t];
  const float bb = beta[head * HDIM + t];

  for (int s = 0; s < SS; ++s) {
    const size_t row = (size_t)(b * SS + s);
    float z  = bfh2f(zbuf[row * NN + head * HDIM + t]);
    float it = gI[row * NHEADS + head];
    float ft = gF[row * NHEADS + head];
    float ot = gO[row * NHEADS + head];

    float mn = fmaxf(ft + m, it);
    float ig = expf(it - mn);
    float fg = expf(ft + m - mn);
    c = fg * c + ig * z;
    n = fg * n + ig;
    m = mn;
    float h = c / (n + 1e-6f) * ot;

    float s1 = h, s2 = h * h;
#pragma unroll
    for (int off = 16; off > 0; off >>= 1) {
      s1 += __shfl_xor(s1, off, 32);
      s2 += __shfl_xor(s2, off, 32);
    }
    if ((t & 31) == 0) { red[wid] = s1; red[4 + wid] = s2; }
    __syncthreads();
    float mu  = (red[0] + red[1] + red[2] + red[3]) * (1.f / HDIM);
    float msq = (red[4] + red[5] + red[6] + red[7]) * (1.f / HDIM);
    __syncthreads();
    float var = fmaxf(msq - mu * mu, 0.f);
    float hn = (h - mu) * rsqrtf(var + 1e-6f) * gg + bb;
    hnbuf[row * NN + head * HDIM + t] = (unsigned short)bf16rne(hn);
  }

  float* co = stateOut;                         // c [B,NH,H]
  float* no = stateOut + BB * NHEADS * HDIM;    // n [B,NH,H]
  float* mo = no + BB * NHEADS * HDIM;          // m [B,NH]
  co[(b * NHEADS + head) * HDIM + t] = c;
  no[(b * NHEADS + head) * HDIM + t] = n;
  if (t == 0) mo[b * NHEADS + head] = m;
}

// =====================================================================
extern "C" void kernel_launch(void* const* d_in, const int* in_sizes, int n_in,
                              void* d_out, int out_size, void* d_ws,
                              size_t ws_size, hipStream_t stream) {
  const float* x      = (const float*)d_in[0];
  const float* conv_w = (const float*)d_in[1];
  const float* conv_b = (const float*)d_in[2];
  const float* Wz     = (const float*)d_in[3];
  const float* Wi     = (const float*)d_in[4];
  const float* Wf     = (const float*)d_in[5];
  const float* Wo     = (const float*)d_in[6];
  const float* gamma  = (const float*)d_in[7];
  const float* beta   = (const float*)d_in[8];
  const float* Wout   = (const float*)d_in[9];

  char* ws = (char*)d_ws;
  unsigned short* zbuf  = (unsigned short*)ws;
  unsigned short* hnbuf = (unsigned short*)(ws + (size_t)MM * NN * 2);
  float* gI = (float*)(ws + (size_t)MM * NN * 4);
  float* gF = gI + (size_t)MM * NHEADS;
  float* gO = gF + (size_t)MM * NHEADS;

  float* out      = (float*)d_out;
  float* stateOut = out + (size_t)MM * NN;  // c, n, m appended flat

  gates_kernel<<<BB * (SS / STILE), GT, 0, stream>>>(
      x, conv_w, conv_b, Wi, Wf, Wo, gI, gF, gO);

  wmma_gemm<true, true><<<dim3(NN / BN, MM / BM), 256, 0, stream>>>(
      (const void*)x, Wz, (void*)zbuf);

  scan_kernel<<<BB * NHEADS, HDIM, 0, stream>>>(
      zbuf, gI, gF, gO, gamma, beta, hnbuf, stateOut);

  wmma_gemm<false, false><<<dim3(NN / BN, MM / BM), 256, 0, stream>>>(
      (const void*)hnbuf, Wout, (void*)out);
}
